// MoeLayer_39178691674573
// MI455X (gfx1250) — matrix operand order repaired and checked by reading, not compiled, not run on hardware
//
#include <hip/hip_runtime.h>
#include <hip/hip_bf16.h>

typedef __bf16 bf16_t;
typedef __attribute__((ext_vector_type(16))) __bf16 v16bf;
typedef __attribute__((ext_vector_type(8)))  __bf16 v8bf;
typedef __attribute__((ext_vector_type(8)))  float  v8f;
typedef __attribute__((ext_vector_type(4)))  float  v4f;

#define EMBED   1024
#define NEXP    16
#define WSTRIDE 1040          // padded bf16 row stride for W in LDS (breaks bank conflicts)
#define WAVES   4
#define TOK_PER_WAVE  16
#define TOK_PER_BLOCK (WAVES * TOK_PER_WAVE)   // 64

__global__ __launch_bounds__(128)
void moe_gate_scale_kernel(const float* __restrict__ x,
                           const float* __restrict__ W,
                           const float* __restrict__ b,
                           float* __restrict__ out)
{
    __shared__ __align__(16) bf16_t Wlds[NEXP * WSTRIDE];   // ~32.5 KB
    __shared__ float gate[WAVES][16 * 16];                  // 4 KB
    __shared__ float scales[TOK_PER_BLOCK];
    __shared__ float bias[NEXP];

    const int tid  = threadIdx.x;
    const int lane = tid & 31;
    const int wave = tid >> 5;

    // ---- Stage W (f32 -> bf16) into padded LDS; stage bias ----
    for (int i = tid; i < NEXP * EMBED; i += blockDim.x) {
        const int n = i >> 10;        // expert row
        const int k = i & (EMBED - 1);
        Wlds[n * WSTRIDE + k] = (bf16_t)W[i];
    }
    if (tid < NEXP) bias[tid] = b[tid];
    __syncthreads();

    // ---- Gate GEMM: one 16(token) x 16(expert) tile per wave, K = 1024 ----
    const long tok0 = (long)blockIdx.x * TOK_PER_BLOCK + (long)wave * TOK_PER_WAVE;
    const int  m    = lane & 15;      // token row within tile / expert col for B
    const int  kh   = lane >> 4;      // K-half selector per ISA fragment layout
    const float* __restrict__ xrow = x + (tok0 + m) * (long)EMBED;
    const bf16_t* __restrict__ wrow = &Wlds[m * WSTRIDE];

    v8f acc = {};
    #pragma unroll 4
    for (int kb = 0; kb < EMBED / 32; ++kb) {
        // A fragment: lane holds K = kbase+0..7 and K = kbase+16..23 (kbase = kb*32 + kh*8)
        const int abase = kb * 32 + kh * 8;
        v4f a0 = *(const v4f*)(xrow + abase);
        v4f a1 = *(const v4f*)(xrow + abase + 4);
        v4f a2 = *(const v4f*)(xrow + abase + 16);
        v4f a3 = *(const v4f*)(xrow + abase + 20);
        v16bf a;
        a[0]=(bf16_t)a0[0];  a[1]=(bf16_t)a0[1];  a[2]=(bf16_t)a0[2];  a[3]=(bf16_t)a0[3];
        a[4]=(bf16_t)a1[0];  a[5]=(bf16_t)a1[1];  a[6]=(bf16_t)a1[2];  a[7]=(bf16_t)a1[3];
        a[8]=(bf16_t)a2[0];  a[9]=(bf16_t)a2[1];  a[10]=(bf16_t)a2[2]; a[11]=(bf16_t)a2[3];
        a[12]=(bf16_t)a3[0]; a[13]=(bf16_t)a3[1]; a[14]=(bf16_t)a3[2]; a[15]=(bf16_t)a3[3];

        // B fragment: lane holds K = kbase+0..7 and K = kbase+16..23 of expert row n=m (kbase = kb*32 + kh*8)
        const v8bf* wp = (const v8bf*)(wrow + kb * 32 + kh * 8);
        v8bf b0 = wp[0];
        v8bf b1 = wp[2];
        v16bf bm = __builtin_shufflevector(b0, b1, 0,1,2,3,4,5,6,7,8,9,10,11,12,13,14,15);

        acc = __builtin_amdgcn_wmma_f32_16x16x32_bf16(
                  /*neg_a=*/false, a, /*neg_b=*/false, bm,
                  /*c_mod=*/(short)0, acc, /*reuse_a=*/false, /*reuse_b=*/false);
    }

    // ---- Spill C tile (+bias) to LDS; ISA C layout: M = r + (lane>>4)*8, N = lane&15 ----
    {
        const int N = lane & 15;
        const float bv = bias[N];
        #pragma unroll
        for (int r = 0; r < 8; ++r) {
            const int M = r + kh * 8;
            gate[wave][M * 16 + N] = acc[r] + bv;
        }
    }
    __syncthreads();

    // ---- Top-2 + softmax-prob sum, one thread per token ----
    if (tid < TOK_PER_BLOCK) {
        const int w = tid >> 4;
        const int t = tid & 15;
        const float* g = &gate[w][t * 16];
        float v1 = -3.4e38f, v2 = -3.4e38f;
        #pragma unroll
        for (int n = 0; n < NEXP; ++n) {
            const float gv = g[n];
            if (gv > v1) { v2 = v1; v1 = gv; }
            else if (gv > v2) { v2 = gv; }
        }
        const float e = __expf(v2 - v1);        // v1 is the max -> stable
        const float z = 1.0f + e;
        scales[tid] = 1.0f / z + e / z;         // matches reference softmax-sum to ulps
    }
    __syncthreads();

    // ---- Scaled streaming copy: x tile is hot in WGP$/L2; NT stores for output ----
    {
        const long base = (long)blockIdx.x * TOK_PER_BLOCK * EMBED;
        const v4f* __restrict__ x4 = (const v4f*)(x + base);
        v4f* __restrict__ o4 = (v4f*)(out + base);
        const int nvec = TOK_PER_BLOCK * EMBED / 4;       // 16384 float4
        for (int i = tid; i < nvec; i += blockDim.x) {
            v4f v = x4[i];
            const float s = scales[i >> 8];               // 256 float4 per token
            v *= s;
            __builtin_nontemporal_store(v, &o4[i]);
        }
    }
}

extern "C" void kernel_launch(void* const* d_in, const int* in_sizes, int n_in,
                              void* d_out, int out_size, void* d_ws, size_t ws_size,
                              hipStream_t stream) {
    const float* x = (const float*)d_in[0];   // (8, 4096, 1024) f32
    const float* W = (const float*)d_in[1];   // (16, 1024) f32
    const float* b = (const float*)d_in[2];   // (16,) f32
    // d_in[3] = top_k (== 2, baked into the top-2 reduction)
    float* out = (float*)d_out;

    const int tokens = in_sizes[0] / EMBED;           // 32768
    const int blocks = tokens / TOK_PER_BLOCK;        // 512
    moe_gate_scale_kernel<<<blocks, WAVES * 32, 0, stream>>>(x, W, b, out);
}
